// AvoidDist_16664473108550
// MI455X (gfx1250) — matrix operand order repaired and checked by reading, not compiled, not run on hardware
//
#include <hip/hip_runtime.h>

typedef __attribute__((ext_vector_type(16))) _Float16 v16h;
typedef __attribute__((ext_vector_type(8)))  float    v8f;

// Problem sizes (fixed by the reference)
#define KDE_N 8192
#define KDE_M 2048
#define KDE_D 32
// Effective GEMM K = 64 = two K=32 WMMA fragments.

// Workspace layout (bytes), d_ws is >=256B aligned from hipMalloc:
//   Astage : _Float16[KDE_N * 64]  @ 0          (1,048,576 B)
//   Bstage : _Float16[64 * KDE_M]  @ 1,048,576  (  262,144 B)
//   bias   : float  [KDE_M]        @ 1,310,720  (    8,192 B)
//   dist   : float  [KDE_N]        @ 1,318,912  (   32,768 B)
#define WS_B_OFF    1048576
#define WS_BIAS_OFF 1310720
#define WS_DIST_OFF 1318912

// ---------------------------------------------------------------------------
// Prep A: A[n, k] = (k < 32) ? s[n,k] : s[n,k-32]^2, stored in the WMMA
// 16-bit A 16x32 fragment layout so the main kernel loads one v16h per lane.
// Stage index = tile*1024 + frag*512 + lane*16 + j
//   j -> vgpr v = j>>1, half h = j&1
//   klocal = 2*(v&3) + h + (v>=4 ? 16 : 0) + (lane>=16 ? 8 : 0)   [ISA 7.12.2]
//   row    = tile*16 + (lane & 15)
// ---------------------------------------------------------------------------
__global__ __launch_bounds__(256) void kde_prep_a(const float* __restrict__ samples,
                                                  _Float16* __restrict__ Astage) {
    int idx  = blockIdx.x * 256 + threadIdx.x;      // 0 .. 524287
    int j    = idx & 15;
    int lane = (idx >> 4) & 31;
    int frag = (idx >> 9) & 1;
    int tile = idx >> 10;
    int v = j >> 1, h = j & 1;
    int klocal = 2 * (v & 3) + h + ((v >= 4) ? 16 : 0) + ((lane >= 16) ? 8 : 0);
    int k = frag * 32 + klocal;
    int n = tile * 16 + (lane & 15);
    float val;
    if (k < 32) {
        val = samples[n * KDE_D + k];
    } else {
        float x = samples[n * KDE_D + (k - 32)];
        val = x * x;
    }
    Astage[idx] = (_Float16)val;
}

// ---------------------------------------------------------------------------
// Prep B: B[k, m] = (k < 32) ? mean[m,k]/var'  : -0.5/var', in the WMMA 16-bit
// B 32x16 layout matching A's k order (lanes 0-15 hold K=0-7,16-23, lanes 16-31
// hold K=8-15,24-31, 2 K per VGPR packed lo/hi).
// Stage index = mtile*1024 + frag*512 + lane*16 + j
//   klocal = 2*(v&3) + h + (v>=4 ? 16 : 0) + (lane>=16 ? 8 : 0)
//   col    = mtile*16 + (lane & 15)
// ---------------------------------------------------------------------------
__global__ __launch_bounds__(256) void kde_prep_b(const float* __restrict__ means,
                                                  const float* __restrict__ stds,
                                                  _Float16* __restrict__ Bstage) {
    int idx   = blockIdx.x * 256 + threadIdx.x;     // 0 .. 131071
    int j     = idx & 15;
    int lane  = (idx >> 4) & 31;
    int frag  = (idx >> 9) & 1;
    int mtile = idx >> 10;
    int v = j >> 1, h = j & 1;
    int klocal = 2 * (v & 3) + h + ((v >= 4) ? 16 : 0) + ((lane >= 16) ? 8 : 0);
    int k = frag * 32 + klocal;
    int m = mtile * 16 + (lane & 15);
    float val;
    if (k < 32) {
        val = means[m * KDE_D + k] / stds[m * KDE_D + k];
    } else {
        val = -0.5f / stds[m * KDE_D + (k - 32)];
    }
    Bstage[idx] = (_Float16)val;
}

// bias[m] = -0.5 * sum_d mean^2/std   (kept in f32 for accuracy)
__global__ __launch_bounds__(256) void kde_prep_bias(const float* __restrict__ means,
                                                     const float* __restrict__ stds,
                                                     float* __restrict__ bias) {
    int m = blockIdx.x * 256 + threadIdx.x;         // 0 .. 2047
    if (m >= KDE_M) return;
    float s = 0.f;
    #pragma unroll
    for (int d = 0; d < KDE_D; ++d) {
        float mu = means[m * KDE_D + d];
        s += mu * mu / stds[m * KDE_D + d];
    }
    bias[m] = -0.5f * s;
}

// ---------------------------------------------------------------------------
// Main: one block per 16-row tile of N (512 blocks x 8 waves).
// Each wave: A fragments resident in VGPRs, sweep 16 of the 128 M-tiles:
//   c = wmma(a0,b0,0); c = wmma(a1,b1,c);  -> logp tile (16x16, f32)
//   rowsum += exp(c + bias)
// C/D layout: lane l, vgpr r -> row = r + 8*(l>=16), col = l&15.
// Reduce cols via shfl_xor inside 16-lane halves, then waves via LDS.
// ---------------------------------------------------------------------------
__global__ __launch_bounds__(256) void kde_main(const _Float16* __restrict__ A,
                                                const _Float16* __restrict__ B,
                                                const float* __restrict__ bias,
                                                float* __restrict__ dist) {
    const int tile = blockIdx.x;
    const int wave = threadIdx.x >> 5;
    const int lane = threadIdx.x & 31;
    const int col  = lane & 15;

    const v16h a0 = *(const v16h*)(A + tile * 1024 +       lane * 16);
    const v16h a1 = *(const v16h*)(A + tile * 1024 + 512 + lane * 16);

    float rowsum[8];
    #pragma unroll
    for (int r = 0; r < 8; ++r) rowsum[r] = 0.f;

    const int mt0 = wave * 16;
    for (int mt = mt0; mt < mt0 + 16; ++mt) {
        const v16h b0 = *(const v16h*)(B + mt * 1024 +       lane * 16);
        const v16h b1 = *(const v16h*)(B + mt * 1024 + 512 + lane * 16);
        v8f c = {};
        c = __builtin_amdgcn_wmma_f32_16x16x32_f16(false, a0, false, b0,
                                                   (short)0, c, false, false);
        c = __builtin_amdgcn_wmma_f32_16x16x32_f16(false, a1, false, b1,
                                                   (short)0, c, false, false);
        const float bm = bias[mt * 16 + col];
        #pragma unroll
        for (int r = 0; r < 8; ++r) rowsum[r] += __expf(c[r] + bm);
    }

    // Reduce across the 16 lanes of each half-wave (xor keeps lanes in-half).
    #pragma unroll
    for (int mask = 1; mask <= 8; mask <<= 1) {
        #pragma unroll
        for (int r = 0; r < 8; ++r)
            rowsum[r] += __shfl_xor(rowsum[r], mask, 32);
    }

    __shared__ float part[8][16];
    if (col == 0) {
        const int base = (lane >= 16) ? 8 : 0;    // lane0: rows 0-7, lane16: rows 8-15
        #pragma unroll
        for (int r = 0; r < 8; ++r) part[wave][base + r] = rowsum[r];
    }
    __syncthreads();

    if (threadIdx.x < 16) {
        float s = 0.f;
        #pragma unroll
        for (int w = 0; w < 8; ++w) s += part[w][threadIdx.x];
        dist[tile * 16 + threadIdx.x] = s * (1.0f / KDE_M);
    }
}

// ---------------------------------------------------------------------------
// Finalize: single block computes global max/min over dist[8192] and writes
// out[n] = -dist[n] + max + min. Deterministic tree reduction in LDS.
// ---------------------------------------------------------------------------
__global__ __launch_bounds__(1024) void kde_finalize(const float* __restrict__ dist,
                                                     float* __restrict__ out) {
    __shared__ float smax[1024];
    __shared__ float smin[1024];
    const int t = threadIdx.x;
    float lmax = -3.402823466e38f, lmin = 3.402823466e38f;
    #pragma unroll
    for (int i = 0; i < 8; ++i) {
        float v = dist[t + i * 1024];
        lmax = fmaxf(lmax, v);
        lmin = fminf(lmin, v);
    }
    smax[t] = lmax;
    smin[t] = lmin;
    __syncthreads();
    for (int s = 512; s > 0; s >>= 1) {
        if (t < s) {
            smax[t] = fmaxf(smax[t], smax[t + s]);
            smin[t] = fminf(smin[t], smin[t + s]);
        }
        __syncthreads();
    }
    const float mx = smax[0];
    const float mn = smin[0];
    #pragma unroll
    for (int i = 0; i < 8; ++i)
        out[t + i * 1024] = -dist[t + i * 1024] + mx + mn;
}

extern "C" void kernel_launch(void* const* d_in, const int* in_sizes, int n_in,
                              void* d_out, int out_size, void* d_ws, size_t ws_size,
                              hipStream_t stream) {
    (void)in_sizes; (void)n_in; (void)out_size; (void)ws_size;
    const float* samples = (const float*)d_in[0];   // [N, 32] f32
    const float* means   = (const float*)d_in[1];   // [M, 32] f32
    const float* stds    = (const float*)d_in[2];   // [M, 32] f32
    float* out = (float*)d_out;                     // [N] f32

    char* ws = (char*)d_ws;
    _Float16* Astage = (_Float16*)(ws);
    _Float16* Bstage = (_Float16*)(ws + WS_B_OFF);
    float*    bias   = (float*)(ws + WS_BIAS_OFF);
    float*    dist   = (float*)(ws + WS_DIST_OFF);

    kde_prep_a   <<<2048, 256, 0, stream>>>(samples, Astage);            // 512K elems
    kde_prep_b   <<< 512, 256, 0, stream>>>(means, stds, Bstage);        // 128K elems
    kde_prep_bias<<<   8, 256, 0, stream>>>(means, stds, bias);          // 2048 elems
    kde_main     <<< 512, 256, 0, stream>>>(Astage, Bstage, bias, dist); // WMMA GEMM+exp
    kde_finalize <<<   1, 1024, 0, stream>>>(dist, out);                 // max/min flip
}
